// IALSHAttention_76570676953297
// MI455X (gfx1250) — matrix operand order repaired and checked by reading, not compiled, not run on hardware
//
#include <hip/hip_runtime.h>

// ---------------------------------------------------------------------------
// IALSH attention mask kernel for MI455X (gfx1250, wave32, WMMA).
// out[b,h,s,t] = (Qsum[t] * (<a[s,:64], qk_s[t,:64]> + a[s,64]*e1[t] + a[s,65]*e2[t]) > 0) ? 0 : -1e4
// GEMM part runs on v_wmma_f32_16x16x32_bf16; rank-2 + threshold epilogue in f32 VALU.
// ---------------------------------------------------------------------------

typedef __attribute__((ext_vector_type(8)))  __bf16 v8bf;
typedef __attribute__((ext_vector_type(16))) __bf16 v16bf;
typedef __attribute__((ext_vector_type(8)))  float  v8f;

constexpr int B_ = 2, H_ = 12, S_ = 2048, D_ = 64;
constexpr int BH    = B_ * H_;        // 24
constexpr int ROWS  = BH * S_;        // 49152 flattened (b,h,s) rows
constexpr int NBLK1 = ROWS / 8;       // 6144 blocks of 8 waves (wave-per-row)
constexpr int STILE = S_ / 16;        // 128 tiles per axis
constexpr int NWAVE3 = BH * STILE * (STILE / 16); // 24576 waves (each does 16 tiles)

__device__ __forceinline__ unsigned short f2bf(float f) {
  // round-to-nearest-even f32 -> bf16
  unsigned u = __float_as_uint(f);
  u += 0x7FFFu + ((u >> 16) & 1u);
  return (unsigned short)(u >> 16);
}

// --- Kernel 1: per-row squared norms + per-block max ------------------------
__global__ void k_norm(const float* __restrict__ qk,
                       float* __restrict__ norm2,
                       float* __restrict__ blockMax) {
  __shared__ float smax[8];
  const int lane = threadIdx.x & 31, wid = threadIdx.x >> 5;
  const int row  = blockIdx.x * 8 + wid;
  const float* qr = qk + (size_t)row * D_;
  float2 q = *(const float2*)(qr + lane * 2);
  float s = q.x * q.x + q.y * q.y;
#pragma unroll
  for (int off = 16; off; off >>= 1) s += __shfl_xor(s, off, 32);
  if (lane == 0) { norm2[row] = s; smax[wid] = s; }
  __syncthreads();
  if (threadIdx.x == 0) {
    float m = smax[0];
#pragma unroll
    for (int i = 1; i < 8; ++i) m = fmaxf(m, smax[i]);
    blockMax[blockIdx.x] = m;
  }
}

// --- Kernel 1b: reduce block maxima -> scalar max(norm^2) -------------------
__global__ void k_maxred(const float* __restrict__ blockMax,
                         float* __restrict__ outMax) {
  __shared__ float sm[256];
  float m = 0.0f;  // norms are non-negative
  for (int i = threadIdx.x; i < NBLK1; i += 256) m = fmaxf(m, blockMax[i]);
  sm[threadIdx.x] = m;
  __syncthreads();
  for (int st = 128; st; st >>= 1) {
    if (threadIdx.x < st) sm[threadIdx.x] = fmaxf(sm[threadIdx.x], sm[threadIdx.x + st]);
    __syncthreads();
  }
  if (threadIdx.x == 0) *outMax = sm[0];
}

// --- Kernel 2: scale, bf16-pack operands, Qsum, ext terms -------------------
__global__ void k_pack(const float* __restrict__ qk, const float* __restrict__ a,
                       const float* __restrict__ norm2, const float* __restrict__ maxn2,
                       unsigned* __restrict__ qsbf, unsigned* __restrict__ abf,
                       float* __restrict__ Qsum,
                       float* __restrict__ e1, float* __restrict__ e2,
                       float* __restrict__ a64, float* __restrict__ a65) {
  const int lane = threadIdx.x & 31, wid = threadIdx.x >> 5;
  const int row  = blockIdx.x * 8 + wid;
  const float scale = 0.75f / __builtin_sqrtf(*maxn2);  // U / max||qk||
  const float* qr = qk + (size_t)row * D_;
  const float* ar = a  + (size_t)row * (D_ + 2);
  float2 q  = *(const float2*)(qr + lane * 2);
  float2 av = *(const float2*)(ar + lane * 2);
  const float qsx = q.x * scale, qsy = q.y * scale;
  qsbf[(size_t)row * 32 + lane] = (unsigned)f2bf(qsx)  | ((unsigned)f2bf(qsy)  << 16);
  abf [(size_t)row * 32 + lane] = (unsigned)f2bf(av.x) | ((unsigned)f2bf(av.y) << 16);
  float s = qsx * av.x + qsy * av.y;  // Q_sum partial (f32, scaled qk)
#pragma unroll
  for (int off = 16; off; off >>= 1) s += __shfl_xor(s, off, 32);
  if (lane == 0) {
    Qsum[row] = s;
    const float n2 = norm2[row];       // UNSCALED norm^2, per reference
    e1[row] = 0.5f - n2;               // 0.5 - ||qk||^2
    e2[row] = 0.5f - n2 * n2;          // 0.5 - ||qk||^4
    a64[row] = ar[64];
    a65[row] = ar[65];
  }
}

// --- Kernel 3: bf16 WMMA batched GEMM + rank-2 + sign epilogue --------------
// Wave handles a 16-row s-strip x 16 consecutive t-tiles. A fragments (a-matrix
// rows) are loaded once; B fragments (qk_s rows, used column-wise) per tile.
// Fragment loads are contiguous row slices matching the wave32 VGPR layouts
// from CDNA5 ISA (7.12.2), so no LDS staging is required.
__global__ void k_gemm(const unsigned short* __restrict__ qsbf,
                       const unsigned short* __restrict__ abf,
                       const float* __restrict__ Qsum,
                       const float* __restrict__ e1, const float* __restrict__ e2,
                       const float* __restrict__ a64, const float* __restrict__ a65,
                       float* __restrict__ out) {
  const int lane = threadIdx.x & 31, wid = threadIdx.x >> 5;
  const int w    = blockIdx.x * 8 + wid;
  const int ttg  = w & 7;             // 8 groups of 16 t-tiles
  const int st   = (w >> 3) & 127;    // s-tile
  const int bh   = w >> 10;           // batch*head
  const int half = lane >> 4;         // lane half selects K sub-range
  const int lp   = lane & 15;

  // A operand (16x32 bf16): M = lane%16; e<8 -> K = 8*half + e,
  //                                      e>=8 -> K = 8*half + 16 + (e-8).
  const unsigned short* arow = abf + ((size_t)(bh * S_) + st * 16 + lp) * D_;
  const int k0 = 8 * half;
  v8bf a00 = *(const v8bf*)(arow + k0);
  v8bf a01 = *(const v8bf*)(arow + k0 + 16);
  v8bf a10 = *(const v8bf*)(arow + 32 + k0);
  v8bf a11 = *(const v8bf*)(arow + 32 + k0 + 16);
  v16bf A0 = __builtin_shufflevector(a00, a01, 0,1,2,3,4,5,6,7,8,9,10,11,12,13,14,15);
  v16bf A1 = __builtin_shufflevector(a10, a11, 0,1,2,3,4,5,6,7,8,9,10,11,12,13,14,15);

  // Rank-2 row terms: s = st*16 + 8*half + r for accumulator VGPR r (C layout).
  const int srow0 = bh * S_ + st * 16 + 8 * half;  // flattened (bh,s) row base
  float4 p0 = *(const float4*)(a64 + srow0);
  float4 p1 = *(const float4*)(a64 + srow0 + 4);
  float4 q0 = *(const float4*)(a65 + srow0);
  float4 q1 = *(const float4*)(a65 + srow0 + 4);
  const float a64r[8] = {p0.x, p0.y, p0.z, p0.w, p1.x, p1.y, p1.z, p1.w};
  const float a65r[8] = {q0.x, q0.y, q0.z, q0.w, q1.x, q1.y, q1.z, q1.w};

  for (int i = 0; i < 16; ++i) {
    const int tt = ttg * 16 + i;
    // B operand (32x16 bf16): N = lane%16 = column t; e<8 -> K = 8*half + e,
    //                                      e>=8 -> K = 8*half + 16 + (e-8).
    // qk_s row-major rows serve as B columns (A * B^T).
    const unsigned short* brow =
        qsbf + ((size_t)(bh * S_) + tt * 16 + lp) * D_ + 8 * half;
    v8bf b00 = *(const v8bf*)(brow);
    v8bf b01 = *(const v8bf*)(brow + 16);
    v8bf b10 = *(const v8bf*)(brow + 32);
    v8bf b11 = *(const v8bf*)(brow + 48);
    v16bf B0 = __builtin_shufflevector(b00, b01, 0,1,2,3,4,5,6,7,8,9,10,11,12,13,14,15);        // K 0..31
    v16bf B1 = __builtin_shufflevector(b10, b11, 0,1,2,3,4,5,6,7,8,9,10,11,12,13,14,15);   // K 32..63

    v8f c = {0.f, 0.f, 0.f, 0.f, 0.f, 0.f, 0.f, 0.f};
    c = __builtin_amdgcn_wmma_f32_16x16x32_bf16(false, A0, false, B0,
                                                (short)0, c, false, false);
    c = __builtin_amdgcn_wmma_f32_16x16x32_bf16(false, A1, false, B1,
                                                (short)0, c, false, false);

    const int t  = tt * 16 + lp;
    const int tb = bh * S_ + t;
    const float Qt = Qsum[tb], E1 = e1[tb], E2 = e2[tb];
#pragma unroll
    for (int r = 0; r < 8; ++r) {
      const float pd  = c[r] + a64r[r] * E1 + a65r[r] * E2;
      const float res = Qt * pd;
      const float ov  = (res > 0.0f) ? 0.0f : -10000.0f;
      // 403 MB streamed once -> non-temporal store, keep L2 for operands.
      __builtin_nontemporal_store(ov, out + (size_t)(srow0 + r) * S_ + t);
    }
  }
}

// ---------------------------------------------------------------------------
extern "C" void kernel_launch(void* const* d_in, const int* in_sizes, int n_in,
                              void* d_out, int out_size, void* d_ws, size_t ws_size,
                              hipStream_t stream) {
  (void)in_sizes; (void)n_in; (void)out_size; (void)ws_size;
  const float* qk = (const float*)d_in[0];   // [B,H,S,64] f32
  const float* a  = (const float*)d_in[1];   // [B,H,S,66] f32
  float* out = (float*)d_out;                // [B,H,S,S] f32
  char* ws = (char*)d_ws;

  auto al = [](size_t x) { return (x + 255) & ~(size_t)255; };
  size_t off = 0;
  const size_t o_max = off; off = al(off + sizeof(float));
  const size_t o_bmx = off; off = al(off + (size_t)NBLK1 * 4);
  const size_t o_n2  = off; off = al(off + (size_t)ROWS * 4);
  const size_t o_qs  = off; off = al(off + (size_t)ROWS * D_ * 2);  // qk_s bf16
  const size_t o_ab  = off; off = al(off + (size_t)ROWS * D_ * 2);  // a    bf16
  const size_t o_qsm = off; off = al(off + (size_t)ROWS * 4);
  const size_t o_e1  = off; off = al(off + (size_t)ROWS * 4);
  const size_t o_e2  = off; off = al(off + (size_t)ROWS * 4);
  const size_t o_a64 = off; off = al(off + (size_t)ROWS * 4);
  const size_t o_a65 = off; off = al(off + (size_t)ROWS * 4);  // total ~13.8 MB

  float* maxn2 = (float*)(ws + o_max);
  float* bmax  = (float*)(ws + o_bmx);
  float* n2    = (float*)(ws + o_n2);
  unsigned* qsbf = (unsigned*)(ws + o_qs);
  unsigned* abf  = (unsigned*)(ws + o_ab);
  float* qsm  = (float*)(ws + o_qsm);
  float* e1   = (float*)(ws + o_e1);
  float* e2   = (float*)(ws + o_e2);
  float* a64p = (float*)(ws + o_a64);
  float* a65p = (float*)(ws + o_a65);

  k_norm  <<<NBLK1, 256, 0, stream>>>(qk, n2, bmax);
  k_maxred<<<1,     256, 0, stream>>>(bmax, maxn2);
  k_pack  <<<NBLK1, 256, 0, stream>>>(qk, a, n2, maxn2, qsbf, abf,
                                      qsm, e1, e2, a64p, a65p);
  k_gemm  <<<NWAVE3 / 8, 256, 0, stream>>>((const unsigned short*)qsbf,
                                           (const unsigned short*)abf,
                                           qsm, e1, e2, a64p, a65p, out);
}
